// NeuMesh_43447889166610
// MI455X (gfx1250) — compile-verified, matrix operand order checked by reading, not run on hardware
//
#include <hip/hip_runtime.h>
#include <hip/hip_bf16.h>

typedef __attribute__((ext_vector_type(16))) _Float16 v16h;
typedef __attribute__((ext_vector_type(8)))  _Float16 v8h;
typedef __attribute__((ext_vector_type(8)))  float    v8f;

#define NSAMP 131072
#define MTILE 64
#define AST   264   // LDS activation row stride in halfs (264*2B = 33*16B -> conflict-free b128 lanes)

// ---------------------------------------------------------------------------
// activation helpers
// ---------------------------------------------------------------------------
__device__ __forceinline__ float softplus100f(float x) {
    float t = 100.0f * x;
    if (t > 20.0f) return x;                       // softplus(100x)/100 ~= x for large x
    return __logf(1.0f + __expf(t)) * 0.01f;
}

// ---------------------------------------------------------------------------
// One dense layer (KP x 256, KP a compile-time multiple of 32), WMMA f16->f32.
//   act : LDS activation buffer, MTILE rows x AST stride (f16). Input cols
//         [0,KP) read; output cols [0,256) written after a barrier.
//   W   : f16 weights, row-major (256 out-rows x KP); acts as B = W^T.
//   MODE: 0 = softplus(beta=100), 1 = ReLU
// Wave w (of 8): row-tile r0 = (w>>1)*16, col-half c0 = (w&1)*128 -> 8 tiles.
// All A/B addresses derive from one per-lane base + compile-time immediates.
// B fragments for a k-tile are preloaded into distinct registers so the
// scheduler can clause the loads and overlap them with the WMMA burst.
// ---------------------------------------------------------------------------
template <int KP, int MODE>
__device__ __forceinline__ void dense_layer_wmma(
    _Float16* act, const _Float16* __restrict__ W, const float* __restrict__ bias,
    int tid)
{
    const int lane = tid & 31;
    const int wave = tid >> 5;
    const int r0   = (wave >> 1) * 16;   // 0,16,32,48
    const int c0   = (wave & 1) * 128;   // 0 or 128
    const int lm   = lane & 15;
    const int lh   = lane >> 4;          // 0/1 half selector

    // per-lane bases; every subsequent offset is a compile-time constant
    const _Float16* abase = act + (r0 + lm) * AST + lh * 8;
    const _Float16* wbase = W + (size_t)(c0 + lm) * KP + lh * 8;

    v8f acc[8] = {};

    constexpr int KT = KP >> 5;
    #pragma unroll
    for (int kt = 0; kt < KT; ++kt) {
        // A fragment (16-bit A layout: lane<16 -> K 0..7 & 16..23 of this tile)
        v16h a;
        {
            v8h a0 = *(const v8h*)(abase + kt * 32);
            v8h a1 = *(const v8h*)(abase + kt * 32 + 16);
            #pragma unroll
            for (int e = 0; e < 8; ++e) { a[e] = a0[e]; a[e + 8] = a1[e]; }
        }
        // preload all 8 B fragments (distinct regs -> clause + partial waits)
        v16h b[8];
        #pragma unroll
        for (int j = 0; j < 8; ++j) {
            const _Float16* bp = wbase + j * 16 * KP + kt * 32;   // const offsets
            v8h b0 = *(const v8h*)(bp);
            v8h b1 = *(const v8h*)(bp + 16);
            #pragma unroll
            for (int e = 0; e < 8; ++e) { b[j][e] = b0[e]; b[j][e + 8] = b1[e]; }
        }
        #pragma unroll
        for (int j = 0; j < 8; ++j) {
            acc[j] = __builtin_amdgcn_wmma_f32_16x16x32_f16(
                         false, a, false, b[j], (short)0, acc[j], false, false);
        }
    }
    __syncthreads();   // all waves done reading act for this layer

    // bias + activation + writeback as f16
    #pragma unroll
    for (int j = 0; j < 8; ++j) {
        const int col = c0 + j * 16 + lm;
        const float bv = bias[col];
        #pragma unroll
        for (int v = 0; v < 8; ++v) {
            float x = acc[j][v] + bv;
            x = (MODE == 0) ? softplus100f(x) : (x > 0.0f ? x : 0.0f);
            act[(r0 + lh * 8 + v) * AST + col] = (_Float16)x;
        }
    }
    __syncthreads();
}

// warm next layer's weights toward the WGP while current layer computes
__device__ __forceinline__ void prefetch_weights(const _Float16* W, int halfs, int tid) {
    const int bytes_per_thread = (halfs * 2) / 256;            // 256 threads/block
    __builtin_prefetch((const char*)W + tid * bytes_per_thread, 0, 3);
    __builtin_prefetch((const char*)W + tid * bytes_per_thread + (bytes_per_thread >> 1), 0, 3);
}

// ---------------------------------------------------------------------------
// Weight prep: weight-norm (density branch) + f16 convert + K padding
// ---------------------------------------------------------------------------
__device__ __forceinline__ void wn_row_f16(const float* v, float g, _Float16* o,
                                           int kin, int kp) {
    float ss = 0.0f;
    for (int k = 0; k < kin; ++k) ss += v[k] * v[k];
    const float sc = g * rsqrtf(ss);
    for (int k = 0; k < kin; ++k) o[k] = (_Float16)(v[k] * sc);
    for (int k = kin; k < kp; ++k) o[k] = (_Float16)0.0f;
}

__device__ __forceinline__ void plain_row_f16(const float* w, _Float16* o,
                                              int kin, int kp) {
    for (int k = 0; k < kin; ++k) o[k] = (_Float16)w[k];
    for (int k = kin; k < kp; ++k) o[k] = (_Float16)0.0f;
}

__global__ void __launch_bounds__(64)
prep_weights(const float* __restrict__ v0, const float* __restrict__ g0,
             const float* __restrict__ v1, const float* __restrict__ g1,
             const float* __restrict__ v2, const float* __restrict__ g2,
             const float* __restrict__ vD, const float* __restrict__ gD,
             const float* __restrict__ wc0, const float* __restrict__ wc1,
             const float* __restrict__ wc2,
             _Float16* __restrict__ W0, _Float16* __restrict__ W1,
             _Float16* __restrict__ W2, _Float16* __restrict__ Wc0,
             _Float16* __restrict__ Wc1, _Float16* __restrict__ Wc2,
             float* __restrict__ wD)
{
    const int r = blockIdx.x * blockDim.x + threadIdx.x;
    if (r < 256) {
        wn_row_f16(v0 + (size_t)r * 177, g0[r], W0 + (size_t)r * 192, 177, 192);
    } else if (r < 512) {
        int q = r - 256;
        wn_row_f16(v1 + (size_t)q * 256, g1[q], W1 + (size_t)q * 256, 256, 256);
    } else if (r < 768) {
        int q = r - 512;
        wn_row_f16(v2 + (size_t)q * 256, g2[q], W2 + (size_t)q * 256, 256, 256);
    } else if (r == 768) {
        float ss = 0.0f;
        for (int k = 0; k < 256; ++k) ss += vD[k] * vD[k];
        const float sc = gD[0] * rsqrtf(ss);
        for (int k = 0; k < 256; ++k) wD[k] = vD[k] * sc;   // keep head in f32
    } else if (r < 1025) {
        int q = r - 769;
        plain_row_f16(wc0 + (size_t)q * 204, Wc0 + (size_t)q * 224, 204, 224);
    } else if (r < 1281) {
        int q = r - 1025;
        plain_row_f16(wc1 + (size_t)q * 256, Wc1 + (size_t)q * 256, 256, 256);
    } else if (r < 1537) {
        int q = r - 1281;
        plain_row_f16(wc2 + (size_t)q * 256, Wc2 + (size_t)q * 256, 256, 256);
    }
}

// ---------------------------------------------------------------------------
// Density branch: gather+embed -> 3 WMMA layers (softplus100) -> 256->1 head
// Input row layout: [d_emb(17) | fg_emb(160) | zeros -> 192]
// ---------------------------------------------------------------------------
__global__ void __launch_bounds__(256)
density_mlp(const float* __restrict__ dcoord, const int* __restrict__ indices,
            const float* __restrict__ bary,  const float* __restrict__ gfeat,
            const _Float16* __restrict__ W0, const float* __restrict__ b0,
            const _Float16* __restrict__ W1, const float* __restrict__ b1,
            const _Float16* __restrict__ W2, const float* __restrict__ b2,
            const float* __restrict__ wD, const float* __restrict__ bD,
            float* __restrict__ out)
{
    __shared__ __align__(16) _Float16 act[MTILE * AST];
    const int tid = threadIdx.x;
    const int n0  = blockIdx.x * MTILE;
    const int s   = tid >> 2;            // sample within tile (0..63)
    const int p   = tid & 3;             // 8-component chunk of the 32-d feature
    const int n   = n0 + s;

    // ---- stage embedded input into LDS ----
    {
        const int   i0 = indices[3 * n + 0], i1 = indices[3 * n + 1], i2 = indices[3 * n + 2];
        const float wa = bary[3 * n + 0],    wb = bary[3 * n + 1],    wc = bary[3 * n + 2];
        const float* ga = gfeat + (size_t)i0 * 32 + p * 8;
        const float* gb = gfeat + (size_t)i1 * 32 + p * 8;
        const float* gc = gfeat + (size_t)i2 * 32 + p * 8;
        _Float16* row = act + s * AST;
        #pragma unroll
        for (int c = 0; c < 8; ++c) {
            const float x  = wa * ga[c] + wb * gb[c] + wc * gc[c];
            const int   cc = p * 8 + c;
            row[17 + cc] = (_Float16)x;                       // fg
            float f = 1.0f;
            #pragma unroll
            for (int k = 0; k < 2; ++k) {                     // L_FG = 2
                row[17 + 32 * (1 + 2 * k) + cc] = (_Float16)__sinf(f * x);
                row[17 + 32 * (2 + 2 * k) + cc] = (_Float16)__cosf(f * x);
                f *= 2.0f;
            }
        }
        if (p == 0) {                                         // d_emb (17)
            const float x = dcoord[n];
            row[0] = (_Float16)x;
            float f = 1.0f;
            #pragma unroll
            for (int k = 0; k < 8; ++k) {                     // L_D = 8
                row[1 + 2 * k] = (_Float16)__sinf(f * x);
                row[2 + 2 * k] = (_Float16)__cosf(f * x);
                f *= 2.0f;
            }
        } else if (p == 1) {
            for (int c = 177; c < 192; ++c) row[c] = (_Float16)0.0f;   // K pad
        }
    }
    prefetch_weights(W0, 256 * 192, tid);
    __syncthreads();

    prefetch_weights(W1, 256 * 256, tid);
    dense_layer_wmma<192, 0>(act, W0, b0, tid);
    prefetch_weights(W2, 256 * 256, tid);
    dense_layer_wmma<256, 0>(act, W1, b1, tid);
    dense_layer_wmma<256, 0>(act, W2, b2, tid);

    // ---- 256 -> 1 head (f32, no activation) ----
    if (tid < MTILE) {
        const _Float16* row = act + tid * AST;
        float acc = bD[0];
        #pragma unroll
        for (int k8 = 0; k8 < 32; ++k8) {
            v8h h = *(const v8h*)(row + k8 * 8);
            #pragma unroll
            for (int e = 0; e < 8; ++e) acc += (float)h[e] * wD[k8 * 8 + e];
        }
        out[n0 + tid] = acc;
    }
}

// ---------------------------------------------------------------------------
// Color branch: gather+embed -> 3 WMMA layers (ReLU) -> 256->3 sigmoid head
// Input row layout: [d_emb(17) | view_emb(27) | ft_emb(160) | zeros -> 224]
// ---------------------------------------------------------------------------
__global__ void __launch_bounds__(256)
color_mlp(const float* __restrict__ dcoord, const float* __restrict__ viewd,
          const int* __restrict__ indices,  const float* __restrict__ bary,
          const float* __restrict__ cfeat,
          const _Float16* __restrict__ W0, const float* __restrict__ b0,
          const _Float16* __restrict__ W1, const float* __restrict__ b1,
          const _Float16* __restrict__ W2, const float* __restrict__ b2,
          const float* __restrict__ wcw, const float* __restrict__ bcw,
          float* __restrict__ out)
{
    __shared__ __align__(16) _Float16 act[MTILE * AST];
    const int tid = threadIdx.x;
    const int n0  = blockIdx.x * MTILE;
    const int s   = tid >> 2;
    const int p   = tid & 3;
    const int n   = n0 + s;

    {
        const int   i0 = indices[3 * n + 0], i1 = indices[3 * n + 1], i2 = indices[3 * n + 2];
        const float wa = bary[3 * n + 0],    wb = bary[3 * n + 1],    wc = bary[3 * n + 2];
        const float* ca  = cfeat + (size_t)i0 * 32 + p * 8;
        const float* cb  = cfeat + (size_t)i1 * 32 + p * 8;
        const float* cc_ = cfeat + (size_t)i2 * 32 + p * 8;
        _Float16* row = act + s * AST;
        #pragma unroll
        for (int c = 0; c < 8; ++c) {                         // ft_emb @ 44
            const float x  = wa * ca[c] + wb * cb[c] + wc * cc_[c];
            const int   cc = p * 8 + c;
            row[44 + cc] = (_Float16)x;
            float f = 1.0f;
            #pragma unroll
            for (int k = 0; k < 2; ++k) {                     // L_FT = 2
                row[44 + 32 * (1 + 2 * k) + cc] = (_Float16)__sinf(f * x);
                row[44 + 32 * (2 + 2 * k) + cc] = (_Float16)__cosf(f * x);
                f *= 2.0f;
            }
        }
        if (p == 0) {                                         // d_emb @ 0
            const float x = dcoord[n];
            row[0] = (_Float16)x;
            float f = 1.0f;
            #pragma unroll
            for (int k = 0; k < 8; ++k) {
                row[1 + 2 * k] = (_Float16)__sinf(f * x);
                row[2 + 2 * k] = (_Float16)__cosf(f * x);
                f *= 2.0f;
            }
        } else if (p == 1) {                                  // view_emb @ 17
            float v3[3] = { viewd[3 * n], viewd[3 * n + 1], viewd[3 * n + 2] };
            #pragma unroll
            for (int i = 0; i < 3; ++i) row[17 + i] = (_Float16)v3[i];
            float f = 1.0f;
            #pragma unroll
            for (int k = 0; k < 4; ++k) {                     // L_VIEW = 4
                #pragma unroll
                for (int i = 0; i < 3; ++i) {
                    row[17 + 3 * (1 + 2 * k) + i] = (_Float16)__sinf(f * v3[i]);
                    row[17 + 3 * (2 + 2 * k) + i] = (_Float16)__cosf(f * v3[i]);
                }
                f *= 2.0f;
            }
        } else if (p == 2) {
            for (int c = 204; c < 224; ++c) row[c] = (_Float16)0.0f;   // K pad
        }
    }
    prefetch_weights(W0, 256 * 224, tid);
    __syncthreads();

    prefetch_weights(W1, 256 * 256, tid);
    dense_layer_wmma<224, 1>(act, W0, b0, tid);
    prefetch_weights(W2, 256 * 256, tid);
    dense_layer_wmma<256, 1>(act, W1, b1, tid);
    dense_layer_wmma<256, 1>(act, W2, b2, tid);

    // ---- 256 -> 3 head with sigmoid ----
    if (tid < MTILE) {
        const _Float16* row = act + tid * AST;
        #pragma unroll
        for (int c = 0; c < 3; ++c) {
            const float* w = wcw + c * 256;
            float acc = bcw[c];
            #pragma unroll
            for (int k8 = 0; k8 < 32; ++k8) {
                v8h h = *(const v8h*)(row + k8 * 8);
                #pragma unroll
                for (int e = 0; e < 8; ++e) acc += (float)h[e] * w[k8 * 8 + e];
            }
            out[NSAMP + 3 * (n0 + tid) + c] = 1.0f / (1.0f + __expf(-acc));
        }
    }
}

// ---------------------------------------------------------------------------
extern "C" void kernel_launch(void* const* d_in, const int* in_sizes, int n_in,
                              void* d_out, int out_size, void* d_ws, size_t ws_size,
                              hipStream_t stream) {
    (void)in_sizes; (void)n_in; (void)out_size; (void)ws_size;
    const float* dcoord  = (const float*)d_in[0];
    const float* viewd   = (const float*)d_in[1];
    const int*   indices = (const int*)d_in[2];
    const float* bary    = (const float*)d_in[3];
    const float* gfeat   = (const float*)d_in[4];
    const float* cfeat   = (const float*)d_in[5];
    // pts_params (v, g, b) x3
    const float* v0 = (const float*)d_in[6];  const float* g0 = (const float*)d_in[7];  const float* b0 = (const float*)d_in[8];
    const float* v1 = (const float*)d_in[9];  const float* g1 = (const float*)d_in[10]; const float* b1 = (const float*)d_in[11];
    const float* v2 = (const float*)d_in[12]; const float* g2 = (const float*)d_in[13]; const float* b2 = (const float*)d_in[14];
    // density_param
    const float* vD = (const float*)d_in[15]; const float* gD = (const float*)d_in[16]; const float* bD = (const float*)d_in[17];
    // views_params (w, b) x3
    const float* wc0 = (const float*)d_in[18]; const float* bc0 = (const float*)d_in[19];
    const float* wc1 = (const float*)d_in[20]; const float* bc1 = (const float*)d_in[21];
    const float* wc2 = (const float*)d_in[22]; const float* bc2 = (const float*)d_in[23];
    // color_param
    const float* wcf = (const float*)d_in[24]; const float* bcf = (const float*)d_in[25];

    // workspace carve-up (f16 padded weights + f32 density head)
    char* ws = (char*)d_ws;
    _Float16* W0p  = (_Float16*)ws; ws += (size_t)256 * 192 * sizeof(_Float16);
    _Float16* W1p  = (_Float16*)ws; ws += (size_t)256 * 256 * sizeof(_Float16);
    _Float16* W2p  = (_Float16*)ws; ws += (size_t)256 * 256 * sizeof(_Float16);
    _Float16* Wc0p = (_Float16*)ws; ws += (size_t)256 * 224 * sizeof(_Float16);
    _Float16* Wc1p = (_Float16*)ws; ws += (size_t)256 * 256 * sizeof(_Float16);
    _Float16* Wc2p = (_Float16*)ws; ws += (size_t)256 * 256 * sizeof(_Float16);
    float*    wDp  = (float*)ws;    ws += (size_t)256 * sizeof(float);

    prep_weights<<<(1537 + 63) / 64, 64, 0, stream>>>(
        v0, g0, v1, g1, v2, g2, vD, gD, wc0, wc1, wc2,
        W0p, W1p, W2p, Wc0p, Wc1p, Wc2p, wDp);

    density_mlp<<<NSAMP / MTILE, 256, 0, stream>>>(
        dcoord, indices, bary, gfeat,
        W0p, b0, W1p, b1, W2p, b2, wDp, bD, (float*)d_out);

    color_mlp<<<NSAMP / MTILE, 256, 0, stream>>>(
        dcoord, viewd, indices, bary, cfeat,
        Wc0p, bc0, Wc1p, bc1, Wc2p, bc2, wcf, bcf, (float*)d_out);
}
